// S4Gen_64450279243808
// MI455X (gfx1250) — compile-verified, matrix operand order checked by reading, not run on hardware
//
#include <hip/hip_runtime.h>
#include <hip/hip_bf16.h>
#include <math.h>

// ---------------- problem constants ----------------
#define BATCH   32
#define DIN     100
#define HD      256        // d_model
#define NL      4
#define NH      32         // half modes
#define LSEQ    4096
#define QC      64         // chunk length
#define NCH     64         // #chunks = LSEQ/QC
#define JCOLS   2048       // BATCH * NCH  (columns of per-head GEMMs)
#define N2      131072     // BATCH * LSEQ (columns of 1x1-conv GEMM)

#define LDK64   72         // padded LDS stride for k=64 tiles (144 B rows, 16B aligned)
#define LDK256  264        // padded LDS stride for k=256 tiles (528 B rows, 16B aligned)

typedef __attribute__((ext_vector_type(16))) __bf16 v16bf;
typedef __attribute__((ext_vector_type(8)))  float  v8f;

#define WMMA_BF16(A, Bm, Cm) \
  __builtin_amdgcn_wmma_f32_16x16x32_bf16(false, (A), false, (Bm), (short)0, (Cm), false, false)

// ---------------- bf16 helpers ----------------
__device__ __forceinline__ unsigned short f2bf(float f) {
  union { float f; unsigned u; } x; x.f = f;
  unsigned u = x.u;
  return (unsigned short)((u + 0x7FFFu + ((u >> 16) & 1u)) >> 16);  // RNE
}
__device__ __forceinline__ float bf2f(unsigned short s) {
  union { unsigned u; float f; } x; x.u = ((unsigned)s) << 16;
  return x.f;
}
__device__ __forceinline__ unsigned pack2(float a, float b) {
  return (unsigned)f2bf(a) | ((unsigned)f2bf(b) << 16);
}

union FragU { unsigned short u[16]; v16bf v; };

// Fragment loader for row-contiguous-k LDS tiles (works for A tiles [m][k] and
// transposed B tiles [n][k]).  ISA 7.12.2 bf16 layout: lane%16 selects row,
// lane>=16 adds 8 to the K base; VGPR v<4: K=kb+2v+{0,1}, v>=4: K=16+kb+2(v-4)+{0,1}.
// Both 8-element halves are contiguous -> lowers to ds_load_b128.
__device__ __forceinline__ v16bf load_frag(const unsigned short* t, int ld, int row0, int k0) {
  int lane = threadIdx.x & 31;
  int row = row0 + (lane & 15);
  int kb  = (lane >> 4) << 3;
  FragU f;
#pragma unroll
  for (int e = 0; e < 16; ++e) {
    int vv = e >> 1, hi = e & 1;
    int k = (vv < 4) ? (kb + (vv << 1) + hi) : (16 + kb + ((vv - 4) << 1) + hi);
    f.u[e] = t[row * ld + k0 + k];
  }
  return f.v;
}

// ---------------- async global->LDS copy (gfx1250, ASYNCcnt-tracked) ----------------
// GVS form: LDS address from the vdst VGPR; global address = SGPR base + VGPR offset
__device__ __forceinline__ void async16(unsigned lds_addr, unsigned long long gbase, unsigned off) {
  asm volatile("global_load_async_to_lds_b128 %0, %1, %2 offset:0"
               :: "v"(lds_addr), "v"(off), "s"(gbase)
               : "memory");
}
__device__ __forceinline__ void async_wait() {
  asm volatile("s_wait_asynccnt 0" ::: "memory");
}
__device__ __forceinline__ unsigned lds_addr_of(const void* p) {
  return (unsigned)(unsigned long long)p;
}

// ---------------- 1) encoder: h[b,o,l] = sum_i x[b,i]*enc_w[i,o,l] ----------------
__global__ void enc_kernel(const float* __restrict__ x, const float* __restrict__ w,
                           float* __restrict__ hbuf) {
  __shared__ float xs[BATCH * DIN];
  int tid = threadIdx.x;
  for (int i = tid; i < BATCH * DIN; i += 256) xs[i] = x[i];
  __syncthreads();
  int idx = blockIdx.x * 256 + tid;          // o*4096 + l  (1M total)
  float acc[BATCH];
#pragma unroll
  for (int b = 0; b < BATCH; ++b) acc[b] = 0.f;
  for (int i = 0; i < DIN; ++i) {
    if (i + 4 < DIN) __builtin_prefetch(w + (size_t)(i + 4) * (HD * LSEQ) + idx, 0, 1);
    float wv = w[i * (HD * LSEQ) + idx];
#pragma unroll
    for (int b = 0; b < BATCH; ++b) acc[b] = fmaf(xs[b * DIN + i], wv, acc[b]);
  }
  int o = idx >> 12, l = idx & 4095;
#pragma unroll
  for (int b = 0; b < BATCH; ++b) hbuf[(((b << 8) + o) << 12) + l] = acc[b];
}

// ---------------- 2) S4D kernel material per head: T (Toeplitz), E, F, w^Q ----------------
__global__ void s4dk_kernel(int layer,
                            const float* __restrict__ log_dt, const float* __restrict__ Cp,
                            const float* __restrict__ logA,   const float* __restrict__ Aim,
                            unsigned short* __restrict__ Tm, unsigned short* __restrict__ Em,
                            unsigned short* __restrict__ Fm, float2* __restrict__ wQ) {
  int h = blockIdx.x;
  int t = threadIdx.x;                       // 0..63
  __shared__ float ksh[QC];
  float dtv  = expf(log_dt[layer * HD + h]);
  float kacc = 0.f;
  int base = (layer * HD + h) * NH;
  for (int n = 0; n < NH; ++n) {
    float ar = -expf(logA[base + n]);
    float ai = Aim[base + n];
    float dar = ar * dtv, dai = ai * dtv;
    float er = expf(dar);
    float wc = er * cosf(dai), wsn = er * sinf(dai);       // w = exp(dt*A)
    float c0 = Cp[(base + n) * 2 + 0], c1 = Cp[(base + n) * 2 + 1];
    // Cc_discrete = (c0+i c1)*(w-1)/(ar+i ai)
    float nr = c0 * (wc - 1.f) - c1 * wsn;
    float ni = c0 * wsn + c1 * (wc - 1.f);
    float den = ar * ar + ai * ai;
    float cr = (nr * ar + ni * ai) / den;
    float ci = (ni * ar - nr * ai) / den;
    // k[t] += 2*Re(Cc * w^t)
    float pt = (float)t;
    float e1 = expf(dar * pt), a1 = e1 * cosf(dai * pt), s1 = e1 * sinf(dai * pt);
    kacc += 2.f * (cr * a1 - ci * s1);
    // E rows: w^{Q-1-t} (Re row 2n, Im row 2n+1)
    float pq = (float)(QC - 1 - t);
    float e2 = expf(dar * pq), a2 = e2 * cosf(dai * pq), s2 = e2 * sinf(dai * pq);
    Em[(h * QC + 2 * n) * QC + t]     = f2bf(a2);
    Em[(h * QC + 2 * n + 1) * QC + t] = f2bf(s2);
    // F cols: z = 2*Cc*w^{t+1};  F[t][2n]=Re z, F[t][2n+1]=-Im z
    float pp = (float)(t + 1);
    float e3 = expf(dar * pp), a3 = e3 * cosf(dai * pp), s3 = e3 * sinf(dai * pp);
    float zr = 2.f * (cr * a3 - ci * s3);
    float zi = 2.f * (cr * s3 + ci * a3);
    Fm[(h * QC + t) * QC + 2 * n]     = f2bf(zr);
    Fm[(h * QC + t) * QC + 2 * n + 1] = f2bf(-zi);
    if (t == 0) {
      float e4 = expf(dar * 64.f);
      wQ[h * NH + n] = make_float2(e4 * cosf(dai * 64.f), e4 * sinf(dai * 64.f));
    }
  }
  ksh[t] = kacc;
  __syncthreads();
  for (int tp = 0; tp < QC; ++tp)
    Tm[(h * QC + t) * QC + tp] = (tp <= t) ? f2bf(ksh[t - tp]) : (unsigned short)0;
}

// ---------------- 3) pack u: hbuf[b][h][l] -> Ubf[h][j][t]  (j=b*64+c, l=c*64+t) ----------------
// consecutive threads: consecutive t -> both read and write coalesced
__global__ void pack_kernel(const float* __restrict__ hbuf, unsigned short* __restrict__ Ubf) {
  int g = blockIdx.x * 256 + threadIdx.x;    // over HD*JCOLS*QC
  int t  = g & 63;
  int j  = (g >> 6) & 2047;
  int hc = g >> 17;
  int b = j >> 6, c = j & 63;
  Ubf[g] = f2bf(hbuf[(((b << 8) + hc) << 12) + (c << 6) + t]);
}

// ---------------- 4) P = E_h @ U_h  (64 x 2048 per head) ----------------
__global__ void gemm_pe_kernel(const unsigned short* __restrict__ Em,
                               const unsigned short* __restrict__ Ubf,
                               unsigned short* __restrict__ Pbf) {
  __shared__ unsigned short Et[QC * LDK64];   // A tile [m][k], padded
  __shared__ unsigned short Ut[QC * LDK64];   // B tile transposed [n][k], padded
  int h = blockIdx.y, j0 = blockIdx.x * 64, tid = threadIdx.x;
  unsigned eL = lds_addr_of(Et), uL = lds_addr_of(Ut);
  unsigned long long Eg = (unsigned long long)(Em + (size_t)h * QC * QC);
  unsigned long long Ug = (unsigned long long)(Ubf + ((size_t)h * JCOLS + j0) * QC);
  for (int i = tid; i < 512; i += 128) {      // 64 rows x 8 (16B) segs each
    int r = i >> 3, sg = i & 7;
    async16(eL + (unsigned)(r * (LDK64 * 2) + sg * 16), Eg, (unsigned)(r * 128 + sg * 16));
    async16(uL + (unsigned)(r * (LDK64 * 2) + sg * 16), Ug, (unsigned)(r * 128 + sg * 16));
  }
  async_wait();
  __syncthreads();
  int w = tid >> 5, n0 = w * 16;
  v8f vz = {0.f, 0.f, 0.f, 0.f, 0.f, 0.f, 0.f, 0.f};
  v8f acc[4];
#pragma unroll
  for (int m = 0; m < 4; ++m) acc[m] = vz;
#pragma unroll
  for (int kk = 0; kk < QC; kk += 32) {
    v16bf bfr = load_frag(Ut, LDK64, n0, kk);
#pragma unroll
    for (int m = 0; m < 4; ++m)
      acc[m] = WMMA_BF16(load_frag(Et, LDK64, m * 16, kk), bfr, acc[m]);
  }
  int lane = tid & 31;
  int colg = j0 + n0 + (lane & 15);
  int radd = (lane >> 4) << 3;                // lane>=16 -> rows +8
  unsigned short* pdst = Pbf + ((size_t)h * JCOLS + colg) * QC + radd;
#pragma unroll
  for (int m = 0; m < 4; ++m) {               // 8 consecutive rows -> one b128
    uint4 pk;
    pk.x = pack2(acc[m][0], acc[m][1]);
    pk.y = pack2(acc[m][2], acc[m][3]);
    pk.z = pack2(acc[m][4], acc[m][5]);
    pk.w = pack2(acc[m][6], acc[m][7]);
    *(uint4*)(pdst + m * 16) = pk;
  }
}

// ---------------- 5) chunk-state scan: s(c+1) = w^Q s(c) + P(c) ----------------
__global__ void scan_kernel(const float2* __restrict__ wQ,
                            const unsigned short* __restrict__ Pbf,
                            unsigned short* __restrict__ Sbf) {
  int g = blockIdx.x * 256 + threadIdx.x;    // HD*NH*BATCH
  int b = g & 31, n = (g >> 5) & 31, h = g >> 10;
  float2 wq = wQ[h * NH + n];
  float sr = 0.f, si = 0.f;
  size_t base = ((size_t)h * JCOLS + (b << 6)) * QC + 2 * n;
  const unsigned short* Pb = Pbf + base;
  unsigned short* Sb = Sbf + base;
  for (int c = 0; c < NCH; ++c) {            // j advances by 1 -> +QC halves
    *(unsigned*)(Sb + (size_t)c * QC) = pack2(sr, si);
    unsigned pv = *(const unsigned*)(Pb + (size_t)c * QC);
    float pr = bf2f((unsigned short)(pv & 0xffffu));
    float pi = bf2f((unsigned short)(pv >> 16));
    float nsr = wq.x * sr - wq.y * si + pr;
    float nsi = wq.x * si + wq.y * sr + pi;
    sr = nsr; si = nsi;
  }
}

// ---------------- 6) Y = T_h@U + F_h@S, +Dskip*u, GELU, emit g[n2][h] bf16 ----------------
__global__ void gemm_y_kernel(const unsigned short* __restrict__ Tm,
                              const unsigned short* __restrict__ Fm,
                              const unsigned short* __restrict__ Ubf,
                              const unsigned short* __restrict__ Sbf,
                              const float* __restrict__ hbuf,
                              const float* __restrict__ Dskip,
                              unsigned short* __restrict__ gbf) {
  __shared__ unsigned short Tt[QC * LDK64], Ft[QC * LDK64];
  __shared__ unsigned short Ut[QC * LDK64], St[QC * LDK64];
  int h = blockIdx.y, j0 = blockIdx.x * 64, tid = threadIdx.x;
  unsigned tL = lds_addr_of(Tt), fL = lds_addr_of(Ft);
  unsigned uL = lds_addr_of(Ut), sL = lds_addr_of(St);
  unsigned long long Tg = (unsigned long long)(Tm + (size_t)h * QC * QC);
  unsigned long long Fg = (unsigned long long)(Fm + (size_t)h * QC * QC);
  unsigned long long Ug = (unsigned long long)(Ubf + ((size_t)h * JCOLS + j0) * QC);
  unsigned long long Sg = (unsigned long long)(Sbf + ((size_t)h * JCOLS + j0) * QC);
  for (int i = tid; i < 512; i += 128) {
    int r = i >> 3, sg = i & 7;
    unsigned lo = (unsigned)(r * (LDK64 * 2) + sg * 16);
    unsigned go = (unsigned)(r * 128 + sg * 16);
    async16(tL + lo, Tg, go);
    async16(fL + lo, Fg, go);
    async16(uL + lo, Ug, go);
    async16(sL + lo, Sg, go);
  }
  async_wait();
  __syncthreads();
  int w = tid >> 5, n0 = w * 16;
  v8f vz = {0.f, 0.f, 0.f, 0.f, 0.f, 0.f, 0.f, 0.f};
  v8f acc[4];
#pragma unroll
  for (int m = 0; m < 4; ++m) acc[m] = vz;
#pragma unroll
  for (int kk = 0; kk < QC; kk += 32) {          // intra-chunk Toeplitz
    v16bf bfr = load_frag(Ut, LDK64, n0, kk);
#pragma unroll
    for (int m = 0; m < 4; ++m)
      acc[m] = WMMA_BF16(load_frag(Tt, LDK64, m * 16, kk), bfr, acc[m]);
  }
#pragma unroll
  for (int kk = 0; kk < QC; kk += 32) {          // inter-chunk state expansion
    v16bf bfr = load_frag(St, LDK64, n0, kk);
#pragma unroll
    for (int m = 0; m < 4; ++m)
      acc[m] = WMMA_BF16(load_frag(Ft, LDK64, m * 16, kk), bfr, acc[m]);
  }
  float dsk = Dskip[h];
  int lane = tid & 31;
  int colg = j0 + n0 + (lane & 15);
  int b = colg >> 6, c = colg & 63;
  int radd = (lane >> 4) << 3;
#pragma unroll
  for (int m = 0; m < 4; ++m)
#pragma unroll
    for (int r = 0; r < 8; ++r) {
      int tt = m * 16 + r + radd;
      int l = (c << 6) + tt;
      float u   = hbuf[(((b << 8) + h) << 12) + l];
      float val = acc[m][r] + dsk * u;
      float gel = 0.5f * val * (1.f + erff(val * 0.70710678f));
      gbf[((size_t)((b << 12) + l)) * HD + h] = f2bf(gel);   // [n2][h]
    }
}

// ---------------- 6b) per-layer out_w f32 -> bf16 ----------------
__global__ void cvtw_kernel(const float* __restrict__ w, unsigned short* __restrict__ wb) {
  int g = blockIdx.x * 256 + threadIdx.x;    // 512*256
  wb[g] = f2bf(w[g]);
}

// ---------------- 7) 1x1 conv GEMM + GLU + residual, in-place on h ----------------
__global__ void gemm_out_kernel(const unsigned short* __restrict__ wbf,  // [512][256] bf16
                                const float* __restrict__ out_b,          // + layer*512
                                const unsigned short* __restrict__ gbf,   // [n2][256]
                                float* __restrict__ hbuf) {
  __shared__ unsigned short Aa[16 * LDK256];
  __shared__ unsigned short Ag[16 * LDK256];
  __shared__ unsigned short Bt[64 * LDK256];  // transposed B [n2][k]
  int ot = blockIdx.y;            // 0..15  (o-tile of a-part; gate at o+256)
  int n0 = blockIdx.x * 64;
  int tid = threadIdx.x;
  unsigned aL = lds_addr_of(Aa), gL = lds_addr_of(Ag), bL = lds_addr_of(Bt);
  unsigned long long Wg = (unsigned long long)wbf;
  unsigned long long Gg = (unsigned long long)gbf;
  for (int i = tid; i < 512; i += 128) {      // 16 rows x 32 segs per A part
    int rr = i >> 5, sg = i & 31;
    unsigned lo = (unsigned)(rr * (LDK256 * 2) + sg * 16);
    async16(aL + lo, Wg, (unsigned)(((ot * 16 + rr) * 256) * 2 + sg * 16));
    async16(gL + lo, Wg, (unsigned)(((ot * 16 + rr + 256) * 256) * 2 + sg * 16));
  }
  for (int i = tid; i < 2048; i += 128) {     // 64 rows x 32 segs
    int rr = i >> 5, sg = i & 31;
    async16(bL + (unsigned)(rr * (LDK256 * 2) + sg * 16), Gg,
            (unsigned)(((size_t)(n0 + rr) * 256) * 2 + sg * 16));
  }
  async_wait();
  __syncthreads();
  int w = tid >> 5, ns = w * 16;
  v8f vz = {0.f, 0.f, 0.f, 0.f, 0.f, 0.f, 0.f, 0.f};
  v8f acca = vz, accg = vz;
#pragma unroll
  for (int kk = 0; kk < 256; kk += 32) {
    v16bf bfr = load_frag(Bt, LDK256, ns, kk);
    acca = WMMA_BF16(load_frag(Aa, LDK256, 0, kk), bfr, acca);
    accg = WMMA_BF16(load_frag(Ag, LDK256, 0, kk), bfr, accg);
  }
  int lane = tid & 31;
  int col = n0 + ns + (lane & 15);
  int bb = col >> 12, l = col & 4095;
  int radd = (lane >> 4) << 3;
#pragma unroll
  for (int r = 0; r < 8; ++r) {
    int o  = ot * 16 + r + radd;
    float a  = acca[r] + out_b[o];
    float gt = accg[r] + out_b[o + 256];
    float glu = a / (1.f + expf(-gt));
    int idx = (((bb << 8) + o) << 12) + l;
    hbuf[idx] += glu;                        // residual, exclusive per lane
  }
}

// ---------------- 8) decoder 1x1 conv ----------------
__global__ void dec_kernel(const float* __restrict__ dec_w, const float* __restrict__ dec_b,
                           const float* __restrict__ hbuf, float* __restrict__ out) {
  int g = blockIdx.x * 256 + threadIdx.x;    // BATCH*LSEQ
  int bb = g >> 12, l = g & 4095;
  float s = dec_b[0];
  for (int hc = 0; hc < HD; ++hc)
    s = fmaf(dec_w[hc], hbuf[(((bb << 8) + hc) << 12) + l], s);
  out[g] = s;
}

// ---------------- host launcher ----------------
extern "C" void kernel_launch(void* const* d_in, const int* in_sizes, int n_in,
                              void* d_out, int out_size, void* d_ws, size_t ws_size,
                              hipStream_t stream) {
  (void)in_sizes; (void)n_in; (void)out_size; (void)ws_size;
  const float* x      = (const float*)d_in[0];
  const float* enc_w  = (const float*)d_in[1];
  const float* log_dt = (const float*)d_in[2];
  const float* Cp     = (const float*)d_in[3];
  const float* logA   = (const float*)d_in[4];
  const float* Aim    = (const float*)d_in[5];
  const float* Dskip  = (const float*)d_in[6];
  const float* out_w  = (const float*)d_in[7];
  const float* out_b  = (const float*)d_in[8];
  const float* dec_w  = (const float*)d_in[9];
  const float* dec_b  = (const float*)d_in[10];
  float* out = (float*)d_out;

  char* ws = (char*)d_ws;
  size_t off = 0;
  float* hbuf           = (float*)(ws + off);          off += (size_t)BATCH * HD * LSEQ * 4;   // 128 MB
  unsigned short* Ubf   = (unsigned short*)(ws + off); off += (size_t)HD * JCOLS * QC * 2;     // 64 MB
  unsigned short* Pbf   = (unsigned short*)(ws + off); off += (size_t)HD * JCOLS * QC * 2;     // 64 MB
  unsigned short* Sbf   = (unsigned short*)(ws + off); off += (size_t)HD * JCOLS * QC * 2;     // 64 MB
  unsigned short* gbf   = (unsigned short*)(ws + off); off += (size_t)N2 * HD * 2;             // 64 MB
  unsigned short* Tm    = (unsigned short*)(ws + off); off += (size_t)HD * QC * QC * 2;        // 2 MB
  unsigned short* Em    = (unsigned short*)(ws + off); off += (size_t)HD * QC * QC * 2;        // 2 MB
  unsigned short* Fm    = (unsigned short*)(ws + off); off += (size_t)HD * QC * QC * 2;        // 2 MB
  unsigned short* wbf   = (unsigned short*)(ws + off); off += (size_t)512 * 256 * 2;           // 256 KB
  float2* wQ            = (float2*)(ws + off);         off += (size_t)HD * NH * 8;

  enc_kernel<<<(HD * LSEQ) / 256, 256, 0, stream>>>(x, enc_w, hbuf);

  for (int layer = 0; layer < NL; ++layer) {
    s4dk_kernel<<<HD, QC, 0, stream>>>(layer, log_dt, Cp, logA, Aim, Tm, Em, Fm, wQ);
    pack_kernel<<<(HD * JCOLS * QC) / 256, 256, 0, stream>>>(hbuf, Ubf);
    cvtw_kernel<<<(512 * 256) / 256, 256, 0, stream>>>(out_w + (size_t)layer * 2 * HD * HD, wbf);
    gemm_pe_kernel<<<dim3(JCOLS / 64, HD), 128, 0, stream>>>(Em, Ubf, Pbf);
    scan_kernel<<<(HD * NH * BATCH) / 256, 256, 0, stream>>>(wQ, Pbf, Sbf);
    gemm_y_kernel<<<dim3(JCOLS / 64, HD), 128, 0, stream>>>(Tm, Fm, Ubf, Sbf, hbuf,
                                                            Dskip + layer * HD, gbf);
    gemm_out_kernel<<<dim3(N2 / 64, 16), 128, 0, stream>>>(wbf, out_b + (size_t)layer * 2 * HD,
                                                           gbf, hbuf);
  }

  dec_kernel<<<(BATCH * LSEQ) / 256, 256, 0, stream>>>(dec_w, dec_b, hbuf, out);
}
